// MaskedSelfAttention_20418274525522
// MI455X (gfx1250) — matrix operand order repaired and checked by reading, not compiled, not run on hardware
//
#include <hip/hip_runtime.h>
#include <hip/hip_bf16.h>

// ---------------------------------------------------------------------------
// MaskedSelfAttention for MI455X (gfx1250): all matmuls on v_wmma_f32_16x16x32_f16
// x:[8,2048,1024] f32, W*:[1024,64], out:[8,2048,64] f32
// ---------------------------------------------------------------------------

typedef __attribute__((ext_vector_type(16))) _Float16 v16h;
typedef __attribute__((ext_vector_type(8)))  _Float16 v8h;
typedef __attribute__((ext_vector_type(4)))  _Float16 v4h;
typedef __attribute__((ext_vector_type(8)))  float    v8f;

#define EMB   1024
#define HEAD  64
#define NB    8
#define SEQ   2048
#define ROWS  (NB * SEQ)        /* 16384 */
#define KCH   (EMB / 32)        /* 32 K-chunks in projection */
#define SKCH  (SEQ / 32)        /* 64 K-chunks in P·V */

// workspace layout (bytes)
static constexpr size_t XH_OFF   = 0;                                        // f16 x, row-major
static constexpr size_t WSWZ_OFF = XH_OFF   + (size_t)ROWS * EMB * 2;        // 3*4*32 chunks * 512 halfs
static constexpr size_t QH_OFF   = WSWZ_OFF + (size_t)3 * 4 * KCH * 512 * 2;
static constexpr size_t KH_OFF   = QH_OFF   + (size_t)ROWS * HEAD * 2;
static constexpr size_t VH_OFF   = KH_OFF   + (size_t)ROWS * HEAD * 2;
static constexpr size_t VSWZ_OFF = VH_OFF   + (size_t)ROWS * HEAD * 2;       // 8*4*64 chunks * 512 halfs
// total ~40.4 MB of d_ws

// -------------------------------------------------- x: f32 -> f16 row-major
__global__ void cvt_x_kernel(const float* __restrict__ x, _Float16* __restrict__ xh) {
  size_t i = (size_t)blockIdx.x * blockDim.x + threadIdx.x;   // 4 floats / thread
  const float4 v = ((const float4*)x)[i];
  v4h h;
  h[0] = (_Float16)v.x; h[1] = (_Float16)v.y;
  h[2] = (_Float16)v.z; h[3] = (_Float16)v.w;
  *(v4h*)(xh + i * 4) = h;
}

// ----- W[1024][64] -> B-fragment order: [(m*4+t)*32+c][lane][16 halfs]
// B operand as used here (contiguous k, same as A): lane = N column (lane%16), elem j -> K = (lane/16)*16 + j
__global__ void prep_w_kernel(const float* __restrict__ Wq, const float* __restrict__ Wk,
                              const float* __restrict__ Wv, _Float16* __restrict__ wswz) {
  int tid = blockIdx.x * blockDim.x + threadIdx.x;            // 12288 threads
  int l = tid & 31, c = (tid >> 5) & 31, t = (tid >> 10) & 3, m = tid >> 12;
  const float* W = (m == 0) ? Wq : ((m == 1) ? Wk : Wv);
  int col = t * 16 + (l & 15);
  int kb  = c * 32 + ((l >> 4) * 16);
  v16h o;
#pragma unroll
  for (int j = 0; j < 16; ++j) o[j] = (_Float16)W[(size_t)(kb + j) * HEAD + col];
  *(v16h*)(wswz + (size_t)tid * 16) = o;
}

// ----- V (f16 [16384][64]) -> B-fragment order per (b, ntile, kchunk)
__global__ void swz_v_kernel(const _Float16* __restrict__ vh, _Float16* __restrict__ vswz) {
  int tid = blockIdx.x * blockDim.x + threadIdx.x;            // 65536 threads
  int l = tid & 31, c = (tid >> 5) & 63, t = (tid >> 11) & 3, b = tid >> 13;
  int col = t * 16 + (l & 15);
  int kb  = c * 32 + ((l >> 4) * 16);
  v16h o;
#pragma unroll
  for (int j = 0; j < 16; ++j) o[j] = vh[(size_t)(b * SEQ + kb + j) * HEAD + col];
  *(v16h*)(vswz + (size_t)tid * 16) = o;
}

// -------------------------------------------------- fused QKV projection
// grid (1024, 3): blockIdx.x = 16-row tile, blockIdx.y = matrix (q/k/v)
// 4 waves/block, wave = N-tile (HEAD/16). K loop: 32 x wmma_f32_16x16x32_f16.
__global__ void __launch_bounds__(128) proj_kernel(
    const _Float16* __restrict__ xh, const _Float16* __restrict__ wswz,
    const float* __restrict__ bq, const float* __restrict__ bk, const float* __restrict__ bv,
    _Float16* __restrict__ qh, _Float16* __restrict__ kh, _Float16* __restrict__ vh) {
  const int m    = blockIdx.y;
  const int wave = threadIdx.x >> 5;          // wave32!
  const int lane = threadIdx.x & 31;
  const int half = lane >> 4;
  const int r16  = lane & 15;
  const int rowBase = blockIdx.x * 16;

  const float*    bias = (m == 0) ? bq : ((m == 1) ? bk : bv);
  _Float16*       outp = (m == 0) ? qh : ((m == 1) ? kh : vh);
  const _Float16* wb   = wswz + (size_t)((m * 4 + wave) * KCH) * 512 + (size_t)lane * 16;
  // A operand as used here (contiguous k, same as B): lane row = lane%16; elem j -> K=half*16+j
  const _Float16* xr   = xh + (size_t)(rowBase + r16) * EMB + half * 16;

  v8f acc = {};
  for (int c = 0; c < KCH; ++c) {
    __builtin_prefetch(wb + (size_t)(c + 1) * 512, 0, 1);
    v16h a;
    ((v8h*)&a)[0] = *(const v8h*)(xr + c * 32);
    ((v8h*)&a)[1] = *(const v8h*)(xr + c * 32 + 8);
    v16h bf = *(const v16h*)(wb + (size_t)c * 512);
    acc = __builtin_amdgcn_wmma_f32_16x16x32_f16(false, a, false, bf, (short)0, acc, false, false);
  }
  const int col = wave * 16 + r16;
  const float bv_ = bias[col];
#pragma unroll
  for (int i = 0; i < 8; ++i) {               // C layout: row = i + 8*half, col = lane%16
    int row = rowBase + i + 8 * half;
    outp[(size_t)row * HEAD + col] = (_Float16)(acc[i] + bv_);
  }
}

// -------------------------------------------------- attention
// grid (128, 8): blockIdx.x = query 16-row tile, blockIdx.y = batch. 256 threads = 8 waves.
// LDS: scores f32[16][2048] | exp(P) f16[16][2048] | reductions  (~194 KB of 320 KB/WGP)
__global__ void __launch_bounds__(256) attn_kernel(
    const _Float16* __restrict__ qh, const _Float16* __restrict__ kh,
    const _Float16* __restrict__ vswz, const int* __restrict__ pad,
    float* __restrict__ out) {
  extern __shared__ char smem[];
  float*    sc     = (float*)smem;                     // 131072 B
  _Float16* ph     = (_Float16*)(smem + 131072);       //  65536 B
  float*    red    = (float*)(smem + 196608);          //   1024 B
  float*    red2   = (float*)(smem + 197632);          //   1024 B
  float*    rowsum = (float*)(smem + 198656);          //     64 B

  const int b     = blockIdx.y;
  const int qbase = blockIdx.x * 16;
  const int tid   = threadIdx.x;
  const int wave  = tid >> 5;
  const int lane  = tid & 31;
  const int half  = lane >> 4;
  const int col   = lane & 15;

  // ---- phase 1: scores = Q K^T / 8, causal + padding mask, into LDS
  const _Float16* qr = qh + (size_t)(b * SEQ + qbase + col) * HEAD + half * 16;
  v16h aq0, aq1;
  ((v8h*)&aq0)[0] = *(const v8h*)(qr);
  ((v8h*)&aq0)[1] = *(const v8h*)(qr + 8);
  ((v8h*)&aq1)[0] = *(const v8h*)(qr + 32);
  ((v8h*)&aq1)[1] = *(const v8h*)(qr + 40);

  for (int kt = wave; kt < SEQ / 16; kt += 8) {
    const int kbase = kt * 16;
    // scores-B[K=h][N=kcol] = k[kcol][h]: contiguous in h -> direct v16h loads
    const _Float16* kr = kh + (size_t)(b * SEQ + kbase + col) * HEAD + half * 16;
    v16h bk0 = *(const v16h*)(kr);
    v16h bk1 = *(const v16h*)(kr + 32);
    v8f c8 = {};
    c8 = __builtin_amdgcn_wmma_f32_16x16x32_f16(false, aq0, false, bk0, (short)0, c8, false, false);
    c8 = __builtin_amdgcn_wmma_f32_16x16x32_f16(false, aq1, false, bk1, (short)0, c8, false, false);
    const int kcol = kbase + col;
    const int pd   = pad[b * SEQ + kcol];
#pragma unroll
    for (int i = 0; i < 8; ++i) {
      int qrow = qbase + i + 8 * half;
      float s = c8[i] * 0.125f;                 // 1/sqrt(64)
      if (kcol > qrow) s -= 1e9f;               // causal (matches reference order)
      if (pd == 0)     s = -1e9f;               // padding overrides
      sc[(i + 8 * half) * SEQ + kcol] = s;
    }
  }
  __syncthreads();

  // ---- phase 2: row softmax (16 rows x 16 segments of 128 cols)
  const int row = tid >> 4, seg = tid & 15, cb = seg * 128;
  float mx = -3.0e38f;
  for (int c = 0; c < 128; ++c) mx = fmaxf(mx, sc[row * SEQ + cb + c]);
  red[row * 16 + seg] = mx;
  __syncthreads();
  if (seg == 0) {
    float mm = red[row * 16];
    for (int i = 1; i < 16; ++i) mm = fmaxf(mm, red[row * 16 + i]);
    red[row * 16] = mm;
  }
  __syncthreads();
  const float rmax = red[row * 16];
  float ls = 0.0f;
  for (int c = 0; c < 128; ++c) {
    float e = __expf(sc[row * SEQ + cb + c] - rmax);
    ph[row * SEQ + cb + c] = (_Float16)e;       // unnormalized; 1/sum folded at the end
    ls += e;
  }
  red2[row * 16 + seg] = ls;
  __syncthreads();
  if (seg == 0) {
    float ss = 0.0f;
    for (int i = 0; i < 16; ++i) ss += red2[row * 16 + i];
    rowsum[row] = ss;
  }
  __syncthreads();

  // ---- phase 3: O = P V. wave -> (ntile = wave&3, K-half = wave>>2), 32 wmma each
  const int nt = wave & 3, kh2 = wave >> 2;
  const _Float16* vb = vswz + (size_t)((b * 4 + nt) * SKCH) * 512 + (size_t)lane * 16;
  const _Float16* pr = ph + (lane & 15) * SEQ + half * 16;
  v8f acc = {};
  for (int c = kh2 * 32; c < kh2 * 32 + 32; ++c) {
    __builtin_prefetch(vb + (size_t)(c + 1) * 512, 0, 1);
    v16h a;
    ((v8h*)&a)[0] = *(const v8h*)(pr + c * 32);
    ((v8h*)&a)[1] = *(const v8h*)(pr + c * 32 + 8);
    v16h bf = *(const v16h*)(vb + (size_t)c * 512);
    acc = __builtin_amdgcn_wmma_f32_16x16x32_f16(false, a, false, bf, (short)0, acc, false, false);
  }
  // combine the two K-halves through LDS (reuse score region) and store
  float* osum = sc;
  if (kh2 == 1) {
#pragma unroll
    for (int i = 0; i < 8; ++i) osum[nt * 256 + (i + 8 * half) * 16 + col] = acc[i];
  }
  __syncthreads();
  if (kh2 == 0) {
#pragma unroll
    for (int i = 0; i < 8; ++i) {
      int r = i + 8 * half;
      float o = (acc[i] + osum[nt * 256 + r * 16 + col]) / rowsum[r];
      out[(size_t)(b * SEQ + qbase + r) * HEAD + nt * 16 + col] = o;
    }
  }
}

// ---------------------------------------------------------------------------
extern "C" void kernel_launch(void* const* d_in, const int* in_sizes, int n_in,
                              void* d_out, int out_size, void* d_ws, size_t ws_size,
                              hipStream_t stream) {
  const float* x   = (const float*)d_in[0];
  const int*   pad = (const int*)d_in[1];
  const float* Wq  = (const float*)d_in[2];
  const float* bq  = (const float*)d_in[3];
  const float* Wk  = (const float*)d_in[4];
  const float* bk  = (const float*)d_in[5];
  const float* Wv  = (const float*)d_in[6];
  const float* bv  = (const float*)d_in[7];
  float* out = (float*)d_out;

  char* ws = (char*)d_ws;
  _Float16* xh   = (_Float16*)(ws + XH_OFF);
  _Float16* wswz = (_Float16*)(ws + WSWZ_OFF);
  _Float16* qh   = (_Float16*)(ws + QH_OFF);
  _Float16* kh   = (_Float16*)(ws + KH_OFF);
  _Float16* vh   = (_Float16*)(ws + VH_OFF);
  _Float16* vswz = (_Float16*)(ws + VSWZ_OFF);

  // 1) x -> f16                      (16 M elems, 4/thread)
  cvt_x_kernel<<<(ROWS * EMB / 4) / 256, 256, 0, stream>>>(x, xh);
  // 2) weights -> B-fragment order   (12288 threads)
  prep_w_kernel<<<96, 128, 0, stream>>>(Wq, Wk, Wv, wswz);
  // 3) fused QKV projection          (1024 row-tiles x 3 matrices, 4 waves each)
  proj_kernel<<<dim3(1024, 3), 128, 0, stream>>>(xh, wswz, bq, bk, bv, qh, kh, vh);
  // 4) V -> B-fragment order         (65536 threads)
  swz_v_kernel<<<512, 128, 0, stream>>>(vh, vswz);
  // 5) attention                     (128 q-tiles x 8 batches, 196 KB dynamic LDS)
  attn_kernel<<<dim3(SEQ / 16, NB), 256, 198720, stream>>>(qh, kh, vswz, pad, out);
}
